// Stacked_GRU_GAT_Predictor_62234076119734
// MI455X (gfx1250) — matrix operand order repaired and checked by reading, not compiled, not run on hardware
//
#include <hip/hip_runtime.h>
#include <hip/hip_bf16.h>

// Shapes (compile-time constants from the reference):
// B=8, N=512, F_IN=128, D=64, H=4, K=15
#define Bb 8
#define Nn 512
#define FIN 128
#define Dd 64
#define Hh 4
#define TOPK 15
#define LRELU 0.2f

typedef __attribute__((ext_vector_type(16))) _Float16 v16h;
typedef __attribute__((ext_vector_type(8)))  float    v8f;
typedef __attribute__((ext_vector_type(4)))  int      v4i;

// ---- gfx1250 async global->LDS path (ASYNCcnt), guarded for toolchain support
#if defined(__AMDGCN__) && __has_builtin(__builtin_amdgcn_global_load_async_to_lds_b128)
#define HAS_ASYNC_LDS 1
typedef __attribute__((address_space(1))) v4i* gas128_t;  // global int4*
typedef __attribute__((address_space(3))) v4i* las128_t;  // LDS int4*
#else
#define HAS_ASYNC_LDS 0
#endif

__device__ __forceinline__ void wait_asynccnt0() {
#if __has_builtin(__builtin_amdgcn_s_wait_asynccnt)
  __builtin_amdgcn_s_wait_asynccnt(0);
#elif defined(__AMDGCN__)
  asm volatile("s_wait_asynccnt 0x0" ::: "memory");
#endif
}

// ---------------- WMMA fragment helpers (16x16x32 f16, wave32) ----------------
// A-matrix 16x32 f16 layout: lane L holds row m = L&15; half = L>>4.
//   element e (0..15): k = k0 + half*16 + e  (contiguous)
__device__ __forceinline__ v16h load_a_frag(const float* __restrict__ A,
                                            int row, int lda, int k0, int lane) {
  const int half = lane >> 4;
  const float* p = A + row * lda + k0 + half * 16;
  v16h a;
#pragma unroll
  for (int o = 0; o < 8; ++o) {
    a[o]     = (_Float16)p[o];
    a[o + 8] = (_Float16)p[o + 8];
  }
  return a;
}

// B-matrix 32x16 f16 layout, where B[k][n] = W[n][k] (W row-major, row stride ldk).
// lane L holds column n = L&15; half = L>>4; element e: k = k0 + half*16 + e.
__device__ __forceinline__ v16h load_bT_frag(const float* __restrict__ W,
                                             int ncol, int ldk, int k0, int lane) {
  const int half = lane >> 4;
  const float* p = W + ncol * ldk + k0 + half * 16;
  v16h b;
#pragma unroll
  for (int e = 0; e < 16; ++e) b[e] = (_Float16)p[e];
  return b;
}

// ---------------- Stage 1: Wh = h @ W^T  (4096x128 x 128x256) ----------------
// Output stored as (B,H,N,D): Wh[((b*H+h)*N + i)*D + d]
__global__ __launch_bounds__(128) void wh_gemm_kernel(const float* __restrict__ h,
                                                      const float* __restrict__ W,
                                                      float* __restrict__ Wh) {
  const int wave = threadIdx.x >> 5;
  const int lane = threadIdx.x & 31;
  const int tile = blockIdx.x * 4 + wave;     // 4096 tiles total
  const int mtile = tile >> 4;                // 256 M tiles (M=4096)
  const int ntile = tile & 15;                // 16 N tiles  (N=256)
  const int m0 = mtile * 16, n0 = ntile * 16;
  const int col = lane & 15;
  const int half = lane >> 4;

  v8f c = {};
#pragma unroll
  for (int kk = 0; kk < FIN; kk += 32) {
    v16h a = load_a_frag(h, m0 + col, FIN, kk, lane);
    v16h b = load_bT_frag(W, n0 + col, FIN, kk, lane);
    c = __builtin_amdgcn_wmma_f32_16x16x32_f16(false, a, false, b, (short)0, c,
                                               false, false);
  }
#pragma unroll
  for (int r = 0; r < 8; ++r) {
    const int m = m0 + r + half * 8;          // row in [0,4096) => (b,i)
    const int n = n0 + col;                   // col in [0,256)  => (h,d)
    const int bb = m >> 9, ii = m & 511;
    const int hd = n >> 6, dd = n & 63;
    Wh[((bb * Hh + hd) * Nn + ii) * Dd + dd] = c[r];
  }
}

// ---------------- Stage 2: per-row scalars e1,e2,sp,ss over D=64 ----------------
// rs layout: [0:16384)=e1, [16384:)=e2, [32768:)=sp, [49152:)=ss   (index = bh*N+i)
__global__ __launch_bounds__(256) void row_scalar_kernel(const float* __restrict__ Wh,
                                                         const float* __restrict__ a,
                                                         const float* __restrict__ pw,
                                                         const float* __restrict__ sw,
                                                         float* __restrict__ rs) {
  const int r = blockIdx.x * blockDim.x + threadIdx.x;   // 16384 rows
  const int hh = (r >> 9) & 3;
  const float* wr = Wh + (size_t)r * Dd;
  float e1 = 0.f, e2 = 0.f, sp = 0.f, ss = 0.f;
#pragma unroll 8
  for (int d = 0; d < Dd; ++d) {
    const float v = wr[d];
    e1 += v * a[hh * 128 + d];
    e2 += v * a[hh * 128 + 64 + d];
    sp += v * pw[d];
    ss += v * sw[d];
  }
  rs[r]         = e1;
  rs[16384 + r] = e2;
  rs[32768 + r] = sp;
  rs[49152 + r] = ss;
}

// ---------------- wave32 reductions ----------------
__device__ __forceinline__ float wred_max(float v) {
#pragma unroll
  for (int o = 16; o; o >>= 1) v = fmaxf(v, __shfl_xor(v, o, 32));
  return v;
}
__device__ __forceinline__ float wred_min(float v) {
#pragma unroll
  for (int o = 16; o; o >>= 1) v = fminf(v, __shfl_xor(v, o, 32));
  return v;
}
__device__ __forceinline__ float wred_sum(float v) {
#pragma unroll
  for (int o = 16; o; o >>= 1) v += __shfl_xor(v, o, 32);
  return v;
}

// ---------------- Stage 3: scores + top/bottom-K softmax + sparse attn@Wh ----
// One block per (b,i): 128 threads = 4 waves, wave w handles head w.
// proj/sim rows are head-independent -> staged once in LDS via the gfx1250
// async global->LDS path; quad row needs a VALU gather so it stays regular.
__global__ __launch_bounds__(128) void attn_kernel(const float* __restrict__ proj,
                                                   const float* __restrict__ sim,
                                                   const int*   __restrict__ quad,
                                                   const float* __restrict__ qbias,
                                                   const float* __restrict__ Wh,
                                                   const float* __restrict__ rs,
                                                   float* __restrict__ hp) {
  __shared__ float projrow[Nn];
  __shared__ float simrow[Nn];
  __shared__ float qbrow[Nn];
  __shared__ int   cnt[Hh];
  __shared__ int   jl[Hh][64];
  __shared__ float wl[Hh][64];

  const int bi = blockIdx.x;            // (b,i) in [0, B*N)
  const int bb = bi >> 9;
  const int ii = bi & 511;
  const int tid = threadIdx.x;

  if (tid < Hh) cnt[tid] = 0;
  const int rowoff = (bb * Nn + ii) * Nn;

#if HAS_ASYNC_LDS
  {
    // 128 lanes x b128 = 512 floats per row; ASYNCcnt-tracked, no VGPR round-trip.
    const int j4 = tid * 4;
    __builtin_amdgcn_global_load_async_to_lds_b128(
        (gas128_t)(proj + rowoff + j4), (las128_t)&projrow[j4], 0, 0);
    __builtin_amdgcn_global_load_async_to_lds_b128(
        (gas128_t)(sim + rowoff + j4), (las128_t)&simrow[j4], 0, 0);
  }
#else
  for (int j = tid; j < Nn; j += 128) {
    projrow[j] = proj[rowoff + j];
    simrow[j]  = sim[rowoff + j];
  }
#endif
  for (int j = tid; j < Nn; j += 128) {
    const int q0 = quad[(size_t)(rowoff + j) * 2];
    const int q1 = quad[(size_t)(rowoff + j) * 2 + 1];
    qbrow[j] = qbias[q0 * 4 + q1];
  }
#if HAS_ASYNC_LDS
  wait_asynccnt0();
#endif
  __syncthreads();

  const int hh = tid >> 5;              // head = wave id
  const int lane = tid & 31;
  const int bh = bb * Hh + hh;
  const float e1v = rs[bh * Nn + ii];
  const float spv = rs[32768 + bh * Nn + ii];
  const float ssv = rs[49152 + bh * Nn + ii];
  const float* e2p = rs + 16384 + bh * Nn;

  // Each lane owns 16 scores: j = t*32 + lane (coalesced in j).
  float myv[16];
#pragma unroll
  for (int t = 0; t < 16; ++t) {
    const int j = t * 32 + lane;
    float e = e1v + e2p[j];
    e = (e >= 0.f) ? e : LRELU * e;
    e += spv * projrow[j] + ssv * simrow[j] + qbrow[j];
    myv[t] = e;
  }

  // 15 strict-bound max iterations -> kth largest; save top1 for softmax shift.
  float bnd_hi = 3.4e38f, top1 = -3.4e38f;
#pragma unroll 1
  for (int it = 0; it < TOPK; ++it) {
    float loc = -3.4e38f;
#pragma unroll
    for (int t = 0; t < 16; ++t) {
      const float v = myv[t];
      if (v < bnd_hi) loc = fmaxf(loc, v);
    }
    bnd_hi = wred_max(loc);
    if (it == 0) top1 = bnd_hi;
  }
  // 15 strict-bound min iterations -> kth smallest.
  float bnd_lo = -3.4e38f;
#pragma unroll 1
  for (int it = 0; it < TOPK; ++it) {
    float loc = 3.4e38f;
#pragma unroll
    for (int t = 0; t < 16; ++t) {
      const float v = myv[t];
      if (v > bnd_lo) loc = fminf(loc, v);
    }
    bnd_lo = wred_min(loc);
  }

  // Active = in top-15 or bottom-15; everything else contributes exp(NEG)->0.
  float lsum = 0.f;
#pragma unroll
  for (int t = 0; t < 16; ++t) {
    const float v = myv[t];
    if (v >= bnd_hi || v <= bnd_lo) {
      const float w = __expf(v - top1);
      lsum += w;
      const int pos = atomicAdd(&cnt[hh], 1);
      if (pos < 64) {
        jl[hh][pos] = t * 32 + lane;
        wl[hh][pos] = w;
      }
    }
  }
  const float sumexp = wred_sum(lsum);
  __syncthreads();

  // Sparse attn @ Wh: lane covers d = lane and lane+32 of D=64.
  const int nact = min(cnt[hh], 64);
  const float inv = 1.f / sumexp;
  const float* whb = Wh + (size_t)(bh * Nn) * Dd;
  float acc0 = 0.f, acc1 = 0.f;
  for (int e = 0; e < nact; ++e) {
    const int j = jl[hh][e];
    const float w = wl[hh][e];
    acc0 += w * whb[j * Dd + lane];
    acc1 += w * whb[j * Dd + lane + 32];
  }
  // hp layout (B,N,H*D) for the final GEMM.
  float* o = hp + ((size_t)(bb * Nn + ii)) * (Hh * Dd) + hh * Dd;
  o[lane]      = acc0 * inv;
  o[lane + 32] = acc1 * inv;
}

// ---------------- Stage 4: out = ELU(hp @ Wc^T)  (4096x256 x 256x64) -----------
__global__ __launch_bounds__(128) void out_gemm_kernel(const float* __restrict__ hp,
                                                       const float* __restrict__ Wc,
                                                       float* __restrict__ out) {
  const int wave = threadIdx.x >> 5;
  const int lane = threadIdx.x & 31;
  const int tile = blockIdx.x * 4 + wave;   // 1024 tiles
  const int mtile = tile >> 2;              // 256 (M=4096)
  const int ntile = tile & 3;               // 4   (N=64)
  const int m0 = mtile * 16, n0 = ntile * 16;
  const int col = lane & 15;
  const int half = lane >> 4;

  v8f c = {};
#pragma unroll
  for (int kk = 0; kk < Hh * Dd; kk += 32) {
    v16h a = load_a_frag(hp, m0 + col, Hh * Dd, kk, lane);
    v16h b = load_bT_frag(Wc, n0 + col, Hh * Dd, kk, lane);
    c = __builtin_amdgcn_wmma_f32_16x16x32_f16(false, a, false, b, (short)0, c,
                                               false, false);
  }
#pragma unroll
  for (int r = 0; r < 8; ++r) {
    const int m = m0 + r + half * 8;
    const int n = n0 + col;
    const float x = c[r];
    const float y = (x > 0.f) ? x : (__expf(x) - 1.f);  // ELU, alpha=1
    out[(size_t)m * 64 + n] = y;
  }
}

// ---------------- Host-side launch ----------------
extern "C" void kernel_launch(void* const* d_in, const int* in_sizes, int n_in,
                              void* d_out, int out_size, void* d_ws, size_t ws_size,
                              hipStream_t stream) {
  (void)in_sizes; (void)n_in; (void)out_size; (void)ws_size;
  const float* h    = (const float*)d_in[0];   // (B,N,F_IN)
  const float* proj = (const float*)d_in[1];   // (B,N,N)
  const float* sim  = (const float*)d_in[2];   // (B,N,N)
  const int*   quad = (const int*)  d_in[3];   // (B,N,N,2)
  const float* W    = (const float*)d_in[4];   // (H*D, F_IN) = (256,128)
  const float* a    = (const float*)d_in[5];   // (H, 2D, 1)  = (4,128,1)
  const float* pw   = (const float*)d_in[6];   // (D,1)
  const float* sw   = (const float*)d_in[7];   // (D,1)
  const float* qb   = (const float*)d_in[8];   // (4,4)
  const float* Wc   = (const float*)d_in[9];   // (D, H*D) = (64,256)

  float* ws = (float*)d_ws;
  float* Wh = ws;                              // (B,H,N,D)      : 1,048,576 f32
  float* rs = ws + 1048576;                    // e1/e2/sp/ss    :    65,536 f32
  float* hp = ws + 1048576 + 65536;            // (B,N,H*D)      : 1,048,576 f32
  float* out = (float*)d_out;                  // (B,N,D)        :   262,144 f32

  wh_gemm_kernel   <<<1024, 128, 0, stream>>>(h, W, Wh);
  row_scalar_kernel<<<  64, 256, 0, stream>>>(Wh, a, pw, sw, rs);
  attn_kernel      <<<4096, 128, 0, stream>>>(proj, sim, quad, qb, Wh, rs, hp);
  out_gemm_kernel  <<< 256, 128, 0, stream>>>(hp, Wc, out);
}
